// HierarchicalQuantumEngine_47098611368143
// MI455X (gfx1250) — matrix operand order repaired and checked by reading, not compiled, not run on hardware
//
#include <hip/hip_runtime.h>
#include <hip/hip_bf16.h>

#define DIMH 512
#define NCELL 1024
#define MMIC 8

typedef __attribute__((ext_vector_type(16))) _Float16 v16h;
typedef __attribute__((ext_vector_type(8)))  float    v8f;

// ---------------- max|x| over 512 ----------------
__global__ void k_sigmax(const float* __restrict__ x, float* __restrict__ sigmax) {
    __shared__ float red[256];
    int t = threadIdx.x;
    float v = 0.f;
    for (int i = t; i < DIMH; i += 256) v = fmaxf(v, fabsf(x[i]));
    red[t] = v; __syncthreads();
    for (int s = 128; s > 0; s >>= 1) { if (t < s) red[t] = fmaxf(red[t], red[t + s]); __syncthreads(); }
    if (t == 0) sigmax[0] = red[0];
}

// -------- per-element cos/sin of rotated phase + per-cell sum(amp0) --------
__global__ void k_init(const float* __restrict__ amp0, const float* __restrict__ ph0,
                       const float* __restrict__ pv,
                       float* __restrict__ wc, float* __restrict__ wsn,
                       float* __restrict__ absum) {
    int bc = blockIdx.x;                 // m*1024 + cell
    size_t base = (size_t)bc * DIMH;
    int t = threadIdx.x;
    __shared__ float red[256];
    float asum = 0.f;
    for (int d = t; d < DIMH; d += 256) {
        size_t idx = base + d;
        float th = ph0[idx] + 0.1f * pv[idx];
        float s, c;
        __sincosf(th, &s, &c);
        wc[idx] = c; wsn[idx] = s;
        asum += amp0[idx];
    }
    red[t] = asum; __syncthreads();
    for (int s2 = 128; s2 > 0; s2 >>= 1) { if (t < s2) red[t] += red[t + s2]; __syncthreads(); }
    if (t == 0) absum[bc] = red[0];
}

// ---------------- main micro-engine step: one block per (m,cell) ----------------
__global__ void k_micro(const float* __restrict__ amp0, const float* __restrict__ wc,
                        const float* __restrict__ wsn,
                        const int* __restrict__ nbr_idx, const float* __restrict__ nbr_mask, int K,
                        const float* __restrict__ absum, const float* __restrict__ x,
                        const float* __restrict__ sigmax, const int* __restrict__ stepp,
                        float* __restrict__ fr, float* __restrict__ fi) {
    int bc = blockIdx.x;
    int m = bc >> 10, cell = bc & 1023;
    int t = threadIdx.x;
    __shared__ int   s_idx[32];
    __shared__ float s_msk[32];
    __shared__ int   s_mid;
    __shared__ float s_cnt;
    __shared__ float red[256];
    if (t < K) { s_idx[t] = nbr_idx[cell * K + t]; s_msk[t] = nbr_mask[cell * K + t]; }
    __syncthreads();
    if (t == 0) {
        float cnt = 0.f, best = -1e30f; int mid = s_idx[0];
        for (int k = 0; k < K; k++) {
            cnt += s_msk[k];
            if (s_msk[k] > 0.f) {
                float sc = absum[(m << 10) + s_idx[k]];
                if (sc > best) { best = sc; mid = s_idx[k]; }   // first-max tie rule
            }
        }
        s_cnt = fmaxf(cnt, 1.f); s_mid = mid;
    }
    __syncthreads();

    float t_glob = 0.1f * (float)stepp[0];
    float wvf = 1.0f + 0.02f * __sinf(t_glob + 6.283185307179586f * (float)cell * (1.0f / 1024.0f));
    float scl = 0.1f / (1.0f + 0.1f * (float)cell);
    float smax = sigmax[0] + 1e-8f;
    size_t mbase = (size_t)(m << 10) * DIMH;
    size_t sbase = (size_t)bc * DIMH;
    int mid = s_mid;
    float invc = 0.03f / s_cnt;           // 0.3 * (0.1 * sum) / cnt

    float nr[2], ni[2];
    float mmax = 0.f;
    for (int q = 0; q < 2; q++) {
        int d = t + q * 256;
        size_t idx = sbase + d;
        float ci = wc[idx], si = wsn[idx], ai = amp0[idx];
        float ire = 0.f, iim = 0.f;
        for (int k = 0; k < K; k++) {
            if (s_msk[k] > 0.f) {
                size_t jdx = mbase + ((size_t)s_idx[k] << 9) + d;
                float cj = wc[jdx], sj = wsn[jdx], aj = amp0[jdx];
                float coup = (ci * cj + si * sj) * aj;  // cos(ang_i-ang_j)*|st_j|
                ire += coup * cj; iim += coup * sj;
            }
        }
        float re = 0.7f * ai * ci + invc * ire;
        float im = 0.7f * ai * si + invc * iim;
        // morph = new*conj(src)/|new| scaled by 0.02
        size_t sdx = mbase + ((size_t)mid << 9) + d;
        float as = amp0[sdx], cs2 = wc[sdx], ss2 = wsn[sdx];
        float nn = sqrtf(re * re + im * im);
        float invn = as / fmaxf(nn, 1e-30f);
        float mre = (re * cs2 + im * ss2) * invn;
        float mim = (im * cs2 - re * ss2) * invn;
        re += 0.02f * mre; im += 0.02f * mim;
        re *= wvf; im *= wvf;
        nr[q] = re; ni[q] = im;
        mmax = fmaxf(mmax, sqrtf(re * re + im * im));
    }
    red[t] = mmax; __syncthreads();
    for (int s2 = 128; s2 > 0; s2 >>= 1) { if (t < s2) red[t] = fmaxf(red[t], red[t + s2]); __syncthreads(); }
    float inv2 = 1.0f / (red[0] + 1e-8f);
    for (int q = 0; q < 2; q++) {
        int d = t + q * 256;
        size_t idx = sbase + d;
        float delta = (x[d] / smax) * 0.3141592653589793f * scl;
        float sd, cd;
        __sincosf(delta, &sd, &cd);
        float re = nr[q], im = ni[q];
        fr[idx] = (re * cd - im * sd) * inv2;
        fi[idx] = (re * sd + im * cd) * inv2;
    }
}

// -------- reduce over cells: micro_out[m,d], cohsum re/im[m,d] --------
__global__ void k_reduce(const float* __restrict__ fr, const float* __restrict__ fi,
                         float* __restrict__ mout, float* __restrict__ cre, float* __restrict__ cim) {
    int m = blockIdx.x >> 2, chunk = blockIdx.x & 3;
    int d = chunk * 128 + threadIdx.x;
    size_t mb = (size_t)(m << 10) * DIMH;
    float sa = 0.f, sr = 0.f, si2 = 0.f;
    for (int cell = 0; cell < NCELL; cell++) {
        size_t idx = mb + ((size_t)cell << 9) + d;
        float a = fr[idx], b = fi[idx];
        float mg = sqrtf(a * a + b * b);
        sa += mg;
        if (mg > 0.f) { sr += a / mg; si2 += b / mg; } else { sr += 1.0f; }  // angle(0)=0 -> e^{i0}=1
    }
    mout[m * DIMH + d] = sa * (1.0f / 1024.0f);
    cre[m * DIMH + d] = sr;
    cim[m * DIMH + d] = si2;
}

__global__ void k_coh(const float* __restrict__ cre, const float* __restrict__ cim,
                      float* __restrict__ coh) {
    int m = blockIdx.x, t = threadIdx.x;
    __shared__ float red[256];
    float s = 0.f;
    for (int d = t; d < DIMH; d += 256) {
        float a = cre[m * DIMH + d] * (1.0f / 1024.0f);
        float b = cim[m * DIMH + d] * (1.0f / 1024.0f);
        s += sqrtf(a * a + b * b);
    }
    red[t] = s; __syncthreads();
    for (int k = 128; k > 0; k >>= 1) { if (t < k) red[t] += red[t + k]; __syncthreads(); }
    if (t == 0) coh[m] = red[0] * (1.0f / DIMH);
}

// -------- zero-pad an [Mrows,K] activation to 16 rows --------
__global__ void k_pad16(const float* __restrict__ src, float* __restrict__ dst,
                        int Mrows, int Kd) {
    int row = blockIdx.x;                     // 0..15
    for (int d = threadIdx.x; d < Kd; d += blockDim.x)
        dst[(size_t)row * Kd + d] = (row < Mrows) ? src[(size_t)row * Kd + d] : 0.0f;
}

// -------- WMMA NT gemm: C[M,N] = A16[16,K] * B[N,K]^T + bias, f16 in / f32 acc --------
// A is pre-padded to 16 rows (zeros beyond Mrows) -> unconditional vector loads.
// One 16x16 tile per wave; 4 waves per block. A/B VGPR layouts per CDNA5 ISA 7.12.2:
//   A lane(L): M=L&15, halves e: K = (e<8 ? e : 16+(e-8)) + 8*(L>>4)
//   B lane(L): N=L&15, halves e: K = (e<8 ? e : 16+(e-8)) + 8*(L>>4)
__global__ void k_gemm_nt(const float* __restrict__ A16, const float* __restrict__ B,
                          const float* __restrict__ bias, float* __restrict__ C,
                          int Mrows, int N, int Kd) {
    int wave = threadIdx.x >> 5;
    int lane = threadIdx.x & 31;
    int ntile = blockIdx.x * 4 + wave;
    if (ntile * 16 >= N) return;                 // wave-uniform
    int mrow = lane & 15;
    int hi = (lane >> 4) & 1;
    const float* Arow = A16 + (size_t)mrow * Kd;
    const float* Brow = B + (size_t)(ntile * 16 + mrow) * Kd;
    v8f acc = {0.f, 0.f, 0.f, 0.f, 0.f, 0.f, 0.f, 0.f};
    for (int kb = 0; kb < Kd; kb += 32) {
        const float4* ap0 = (const float4*)(Arow + kb + hi * 8);
        const float4* ap1 = (const float4*)(Arow + kb + 16 + hi * 8);
        const float4* bp0 = (const float4*)(Brow + kb + hi * 8);
        const float4* bp1 = (const float4*)(Brow + kb + 16 + hi * 8);
        float4 a0 = ap0[0], a1 = ap0[1];         // e=0..7   (K = kb+8hi+e)
        float4 a2 = ap1[0], a3 = ap1[1];         // e=8..15  (K = kb+16+8hi+(e-8))
        float4 b0 = bp0[0], b1 = bp0[1], b2 = bp1[0], b3 = bp1[1];  // same K mapping as A
        v16h a, b;
        a[0] = (_Float16)a0.x; a[1] = (_Float16)a0.y; a[2]  = (_Float16)a0.z; a[3]  = (_Float16)a0.w;
        a[4] = (_Float16)a1.x; a[5] = (_Float16)a1.y; a[6]  = (_Float16)a1.z; a[7]  = (_Float16)a1.w;
        a[8] = (_Float16)a2.x; a[9] = (_Float16)a2.y; a[10] = (_Float16)a2.z; a[11] = (_Float16)a2.w;
        a[12] = (_Float16)a3.x; a[13] = (_Float16)a3.y; a[14] = (_Float16)a3.z; a[15] = (_Float16)a3.w;
        b[0] = (_Float16)b0.x; b[1] = (_Float16)b0.y; b[2]  = (_Float16)b0.z; b[3]  = (_Float16)b0.w;
        b[4] = (_Float16)b1.x; b[5] = (_Float16)b1.y; b[6]  = (_Float16)b1.z; b[7]  = (_Float16)b1.w;
        b[8] = (_Float16)b2.x; b[9] = (_Float16)b2.y; b[10] = (_Float16)b2.z; b[11] = (_Float16)b2.w;
        b[12] = (_Float16)b3.x; b[13] = (_Float16)b3.y; b[14] = (_Float16)b3.z; b[15] = (_Float16)b3.w;
        acc = __builtin_amdgcn_wmma_f32_16x16x32_f16(false, a, false, b, (short)0, acc,
                                                     false, false);
    }
    int col = ntile * 16 + mrow;
    float bv = bias ? bias[col] : 0.0f;
#pragma unroll
    for (int r = 0; r < 8; r++) {
        int Mr = r + hi * 8;                     // C/D layout: VGPR r -> M=r (lanes 0-15), M=r+8 (16-31)
        if (Mr < Mrows) C[(size_t)Mr * N + col] = acc[r] + bv;
    }
}

// -------- GRU gates (torch order r,z,n) --------
__global__ void k_gates(const float* __restrict__ gx, const float* __restrict__ gh,
                        const float* __restrict__ h0, float* __restrict__ h) {
    int idx = blockIdx.x * blockDim.x + threadIdx.x;   // 4096
    int i = idx >> 9, j = idx & 511;
    const float* gxr = gx + i * 1536;
    const float* ghr = gh + i * 1536;
    float r = 1.f / (1.f + expf(-(gxr[j] + ghr[j])));
    float z = 1.f / (1.f + expf(-(gxr[512 + j] + ghr[512 + j])));
    float ng = tanhf(gxr[1024 + j] + r * ghr[1024 + j]);
    h[idx] = (1.f - z) * ng + z * h0[idx];
}

// -------- 8x8 attention: scores, coherence gating, softmax, column weights --------
__global__ void k_attn(const float* __restrict__ Q, const float* __restrict__ Km,
                       const float* __restrict__ coh, float* __restrict__ colw) {
    __shared__ float sc[64]; __shared__ float at[64]; __shared__ float rs[8];
    int t = threadIdx.x;            // 64
    int i = t >> 3, j = t & 7;
    float dot = 0.f;
    for (int k2 = 0; k2 < DIMH; k2++) dot += Q[i * DIMH + k2] * Km[j * DIMH + k2];
    float fac = 0.5f + 0.5f * coh[i] * coh[j];
    sc[t] = dot * 0.044194173824159216f * fac;   // 1/sqrt(512)
    __syncthreads();
    if (t < 8) {
        float mx = -1e30f;
        for (int jj = 0; jj < 8; jj++) mx = fmaxf(mx, sc[t * 8 + jj]);
        float ss = 0.f;
        for (int jj = 0; jj < 8; jj++) { float e = expf(sc[t * 8 + jj] - mx); at[t * 8 + jj] = e; ss += e; }
        rs[t] = ss;
    }
    __syncthreads();
    if (t < 8) {
        float cw = 0.f;
        for (int ii = 0; ii < 8; ii++) cw += at[ii * 8 + t] / rs[ii];
        colw[t] = cw * 0.125f;                   // mean over rows folded in
    }
}

__global__ void k_attended(const float* __restrict__ colw, const float* __restrict__ V,
                           float* __restrict__ att) {
    int d = threadIdx.x;            // 512
    float s = 0.f;
    for (int j = 0; j < 8; j++) s += colw[j] * V[j * DIMH + d];
    att[d] = s;
}

// -------- faction sync + unbiased variance (tension) --------
__global__ void k_tension(const float* __restrict__ h, const int* __restrict__ stepp,
                          float* __restrict__ out_t) {
    __shared__ float red[512];
    int d = threadIdx.x;            // 512
    float hv[8], v[8];
    for (int i = 0; i < 8; i++) hv[i] = h[i * DIMH + d];
    float go = 0.f;
    for (int f = 0; f < 4; f++) {
        float pm = 0.5f * (hv[2 * f] + hv[2 * f + 1]);
        v[2 * f]     = 0.85f * hv[2 * f]     + 0.15f * pm;
        v[2 * f + 1] = 0.85f * hv[2 * f + 1] + 0.15f * pm;
        go += pm;
    }
    go *= 0.25f;                    // overall mean of h[:,d]
    if (stepp[0] > 5) {
        for (int f = 0; f < 4; f++) v[2 * f] = 0.85f * v[2 * f] + 0.15f * go;  // dc=1 -> s=0 only
    }
    float s = 0.f;
    for (int i = 0; i < 8; i++) s += v[i];
    red[d] = s; __syncthreads();
    for (int k = 256; k > 0; k >>= 1) { if (d < k) red[d] += red[d + k]; __syncthreads(); }
    float mean = red[0] * (1.0f / 4096.0f);
    __syncthreads();
    float sq = 0.f;
    for (int i = 0; i < 8; i++) { float dd = v[i] - mean; sq += dd * dd; }
    red[d] = sq; __syncthreads();
    for (int k = 256; k > 0; k >>= 1) { if (d < k) red[d] += red[d + k]; __syncthreads(); }
    if (d == 0) out_t[0] = red[0] / 4095.0f;     // unbiased
}

extern "C" void kernel_launch(void* const* d_in, const int* in_sizes, int n_in,
                              void* d_out, int out_size, void* d_ws, size_t ws_size,
                              hipStream_t stream) {
    (void)n_in; (void)out_size; (void)ws_size;
    const float* x        = (const float*)d_in[0];
    const int*   step     = (const int*)  d_in[1];
    const float* amp0     = (const float*)d_in[2];
    const float* ph0      = (const float*)d_in[3];
    const float* pv       = (const float*)d_in[4];
    const int*   nbr_idx  = (const int*)  d_in[5];
    const float* nbr_mask = (const float*)d_in[6];
    const float* h0       = (const float*)d_in[7];
    const float* w_ih     = (const float*)d_in[8];
    const float* b_ih     = (const float*)d_in[9];
    const float* w_hh     = (const float*)d_in[10];
    const float* b_hh     = (const float*)d_in[11];
    const float* wq       = (const float*)d_in[12];
    const float* bq       = (const float*)d_in[13];
    const float* wk       = (const float*)d_in[14];
    const float* bk       = (const float*)d_in[15];
    const float* wv       = (const float*)d_in[16];
    const float* bv       = (const float*)d_in[17];
    const float* out_w    = (const float*)d_in[18];
    const float* out_b    = (const float*)d_in[19];
    float* out = (float*)d_out;

    int K = in_sizes[5] / NCELL;
    if (K > 32) K = 32;

    const size_t NE = (size_t)MMIC * NCELL * DIMH;    // 4,194,304
    float* ws = (float*)d_ws;
    float* w_c    = ws;
    float* w_s    = ws + NE;
    float* w_fr   = ws + 2 * NE;
    float* w_fi   = ws + 3 * NE;
    float* w_abs  = ws + 4 * NE;        // 8192
    float* w_smax = w_abs + MMIC * NCELL;
    float* w_mout = w_smax + 1;         // 4096
    float* w_cre  = w_mout + MMIC * DIMH;
    float* w_cim  = w_cre + MMIC * DIMH;
    float* w_coh  = w_cim + MMIC * DIMH;  // 8
    float* w_gx   = w_coh + MMIC;       // 12288
    float* w_gh   = w_gx + MMIC * 3 * DIMH;
    float* w_h    = w_gh + MMIC * 3 * DIMH; // 4096
    float* w_q    = w_h + MMIC * DIMH;
    float* w_k    = w_q + MMIC * DIMH;
    float* w_v    = w_k + MMIC * DIMH;
    float* w_colw = w_v + MMIC * DIMH;  // 8
    float* w_att  = w_colw + MMIC;      // 512
    float* w_pad  = w_att + DIMH;       // 16*512 padded-A staging

    k_sigmax<<<1, 256, 0, stream>>>(x, w_smax);
    k_init<<<MMIC * NCELL, 256, 0, stream>>>(amp0, ph0, pv, w_c, w_s, w_abs);
    k_micro<<<MMIC * NCELL, 256, 0, stream>>>(amp0, w_c, w_s, nbr_idx, nbr_mask, K,
                                              w_abs, x, w_smax, step, w_fr, w_fi);
    k_reduce<<<MMIC * 4, 128, 0, stream>>>(w_fr, w_fi, w_mout, w_cre, w_cim);
    k_coh<<<MMIC, 256, 0, stream>>>(w_cre, w_cim, w_coh);

    // GRU input/hidden gemms: [8,512] x [1536,512]^T (A zero-padded to 16 rows)
    k_pad16<<<16, 256, 0, stream>>>(w_mout, w_pad, MMIC, DIMH);
    k_gemm_nt<<<(3 * DIMH) / 64, 128, 0, stream>>>(w_pad, w_ih, b_ih, w_gx, MMIC, 3 * DIMH, DIMH);
    k_pad16<<<16, 256, 0, stream>>>(h0, w_pad, MMIC, DIMH);
    k_gemm_nt<<<(3 * DIMH) / 64, 128, 0, stream>>>(w_pad, w_hh, b_hh, w_gh, MMIC, 3 * DIMH, DIMH);
    k_gates<<<8, 512, 0, stream>>>(w_gx, w_gh, h0, w_h);

    // QKV: [8,512] x [512,512]^T (pad h once, reuse for all three)
    k_pad16<<<16, 256, 0, stream>>>(w_h, w_pad, MMIC, DIMH);
    k_gemm_nt<<<DIMH / 64, 128, 0, stream>>>(w_pad, wq, bq, w_q, MMIC, DIMH, DIMH);
    k_gemm_nt<<<DIMH / 64, 128, 0, stream>>>(w_pad, wk, bk, w_k, MMIC, DIMH, DIMH);
    k_gemm_nt<<<DIMH / 64, 128, 0, stream>>>(w_pad, wv, bv, w_v, MMIC, DIMH, DIMH);

    k_attn<<<1, 64, 0, stream>>>(w_q, w_k, w_coh, w_colw);
    k_attended<<<1, 512, 0, stream>>>(w_colw, w_v, w_att);

    // pred: [1,512] x [512,512]^T -> d_out[0..511]
    k_pad16<<<16, 256, 0, stream>>>(w_att, w_pad, 1, DIMH);
    k_gemm_nt<<<DIMH / 64, 128, 0, stream>>>(w_pad, out_w, out_b, out, 1, DIMH, DIMH);

    // tension -> d_out[512]
    k_tension<<<1, 512, 0, stream>>>(w_h, step, out + DIMH);
}
